// PointBEVSampling_41781441855752
// MI455X (gfx1250) — matrix operand order repaired and checked by reading, not compiled, not run on hardware
//
#include <hip/hip_runtime.h>
#include <cstddef>
#include <cstdint>

#define Q_    20000
#define Z_    4
#define C_    128
#define NCAM_ 6
#define QZ_   (Q_ * Z_)

typedef __attribute__((ext_vector_type(16))) __bf16 v16bf;
typedef __attribute__((ext_vector_type(8)))  float  v8f;

static __device__ __forceinline__ unsigned short f2bfu(float f) {
  union { float f; unsigned u; } v; v.f = f;
  unsigned u = v.u;
  u += 0x7fffu + ((u >> 16) & 1u);   // round-to-nearest-even
  return (unsigned short)(u >> 16);
}

// CDNA5 async copy: 32B per lane (two b128 chunks; INST_OFFSET applies to both
// the LDS and global address per ISA 08 §4.4, so the same operands are reused).
static __device__ __forceinline__ void async_ld32(unsigned lds, const void* g) {
  asm volatile("global_load_async_to_lds_b128 %0, %1, off"
               :: "v"(lds), "v"(g) : "memory");
  asm volatile("global_load_async_to_lds_b128 %0, %1, off offset:16"
               :: "v"(lds), "v"(g) : "memory");
}
static __device__ __forceinline__ void wait_async0() {
  asm volatile("s_wait_asynccnt 0" ::: "memory");
}

// ---------------- weight conversion f32 -> bf16 (rows padded K -> Kp) ----------------
__global__ void convert_w_kernel(const float* __restrict__ src,
                                 unsigned short* __restrict__ dst,
                                 int K, int N, int Kp) {
  int i = blockIdx.x * blockDim.x + threadIdx.x;
  if (i >= Kp * N) return;
  int k = i / N, n = i - k * N;
  dst[i] = (k < K) ? f2bfu(src[(size_t)k * N + n]) : (unsigned short)0;
}

// ---------------- sinusoidal PE features -> bf16 [QZ, 64] (60 + 4 zero pad) ----------------
__global__ void pe_feat_kernel(const float* __restrict__ rp,
                               unsigned short* __restrict__ out) {
  int p = blockIdx.x * blockDim.x + threadIdx.x;
  if (p >= QZ_) return;
  int q = p >> 2, z = p & 3;
  alignas(16) unsigned short buf[64];
  const float PI = 3.14159265358979323846f;
#pragma unroll
  for (int d = 0; d < 3; ++d) {
    float x = rp[((size_t)z * Q_ + q) * 3 + d];
    float a = x * PI;                      // freq_f = 2^f * pi
#pragma unroll
    for (int f = 0; f < 10; ++f) {
      buf[d * 20 + f]      = f2bfu(sinf(a));
      buf[d * 20 + 10 + f] = f2bfu(cosf(a));
      a *= 2.0f;
    }
  }
  buf[60] = buf[61] = buf[62] = buf[63] = 0;
  uint4* dst = reinterpret_cast<uint4*>(out + (size_t)p * 64);
  const uint4* s = reinterpret_cast<const uint4*>(buf);
#pragma unroll
  for (int i = 0; i < 8; ++i) dst[i] = s[i];
}

// ---------------- bf16 WMMA GEMM: D[M,N] = act(A[M,K] @ W[K,N] + bias) ----------------
// Block tile 128x128, 8 waves each computing 32x64 (2x4 WMMA 16x16 accums).
// A tiles: async global->LDS (ASYNCcnt), LDS double-buffered, overlapped with WMMA.
// B tiles: sync load + K-pair-packed transposed LDS store (weights are L2-resident).
template <bool RELU, bool OUTBF16>
__global__ __launch_bounds__(256) void gemm_bf16_kernel(
    const __bf16* __restrict__ A, const __bf16* __restrict__ Bw,
    const float* __restrict__ bias, void* __restrict__ Dout,
    int M, int N, int K) {
  constexpr int BM = 128, BN = 128, BK = 32, LDP = BK + 8;  // row stride 80B (16B aligned)
  __shared__ alignas(16) __bf16 As[2][BM][LDP];
  __shared__ alignas(16) __bf16 Bs[2][BN][LDP];   // N-major, K contiguous per column

  const int tid  = threadIdx.x;
  const int lane = tid & 31;
  const int wid  = tid >> 5;
  const int wm   = (wid & 3) * 32;   // wave M offset in tile
  const int wn   = (wid >> 2) * 64;  // wave N offset in tile
  const int mBase = blockIdx.y * BM;
  const int nBase = blockIdx.x * BN;

  const int  l16 = lane & 15;
  const bool hi  = (lane & 16) != 0;
  const int  akb = hi ? 8 : 0;    // A frag: lanes 16-31 hold K=8..15 / 24..31
  const int  bkb = hi ? 16 : 0;   // unused; B frag reads with akb, K=8..15 / 24..31 like A

  v8f zero = {};
  v8f c[2][4];
#pragma unroll
  for (int mi = 0; mi < 2; ++mi)
#pragma unroll
    for (int ni = 0; ni < 4; ++ni) c[mi][ni] = zero;

  const int ar  = tid >> 1;           // A: 2 threads/row, 32B each
  const int ac  = (tid & 1) * 16;
  const int bk2 = (tid >> 4) * 2;     // B: K-row pair 0,2,..,30
  const int bn8 = (tid & 15) * 8;     // B: 8 columns per thread

  auto loadA = [&](int kt, int buf) {
    int gr = mBase + ar;
    if (gr < M) {
      unsigned lds = (unsigned)(uintptr_t)(&As[buf][ar][ac]);
      async_ld32(lds, A + (size_t)gr * K + kt * BK + ac);
    } else {
      uint4 zz = {0, 0, 0, 0};
      *reinterpret_cast<uint4*>(&As[buf][ar][ac])     = zz;
      *reinterpret_cast<uint4*>(&As[buf][ar][ac + 8]) = zz;
    }
  };
  auto loadB = [&](int kt, int buf) {
    const uint4* s0 =
        reinterpret_cast<const uint4*>(Bw + (size_t)(kt * BK + bk2) * N + nBase + bn8);
    const uint4* s1 =
        reinterpret_cast<const uint4*>(Bw + (size_t)(kt * BK + bk2 + 1) * N + nBase + bn8);
    uint4 r0 = s0[0], r1 = s1[0];
    const unsigned short* p0 = reinterpret_cast<const unsigned short*>(&r0);
    const unsigned short* p1 = reinterpret_cast<const unsigned short*>(&r1);
#pragma unroll
    for (int e = 0; e < 8; ++e) {  // pack (k, k+1) pair -> one b32 LDS store
      unsigned v = (unsigned)p0[e] | ((unsigned)p1[e] << 16);
      *reinterpret_cast<unsigned*>(&Bs[buf][bn8 + e][bk2]) = v;
    }
  };

  const int nk = K / BK;
  loadA(0, 0);
  loadB(0, 0);

  for (int kt = 0; kt < nk; ++kt) {
    const int cur = kt & 1;
    wait_async0();        // own tile-kt async chunks done
    __syncthreads();      // => all waves' tile-kt writes visible
    if (kt + 1 < nk) {    // prefetch next tile into other buffer (overlaps WMMA)
      loadA(kt + 1, cur ^ 1);
      loadB(kt + 1, cur ^ 1);
    }

    v16bf a[2], b[4];
#pragma unroll
    for (int j = 0; j < 8; ++j) {   // A 16x32 fragment layout (ISA 7.12.2)
      a[0][j]     = As[cur][wm + l16][akb + j];
      a[0][j + 8] = As[cur][wm + l16][akb + 16 + j];
      a[1][j]     = As[cur][wm + 16 + l16][akb + j];
      a[1][j + 8] = As[cur][wm + 16 + l16][akb + 16 + j];
    }
#pragma unroll
    for (int ni = 0; ni < 4; ++ni)
#pragma unroll
      for (int j = 0; j < 16; ++j)  // B 32x16 fragment layout
        b[ni][j] = Bs[cur][wn + ni * 16 + l16][akb + (j < 8 ? j : j + 8)];

#pragma unroll
    for (int mi = 0; mi < 2; ++mi)
#pragma unroll
      for (int ni = 0; ni < 4; ++ni)
        c[mi][ni] = __builtin_amdgcn_wmma_f32_16x16x32_bf16(
            false, a[mi], false, b[ni], (short)0, c[mi][ni], false, false);
  }

  // -------- epilogue: block-uniform fast path when the whole M-tile is in range --------
  const int rb = hi ? 8 : 0;   // C/D layout: lanes 16-31 hold rows M=i+8
  if (mBase + BM <= M) {
#pragma unroll
    for (int mi = 0; mi < 2; ++mi)
#pragma unroll
      for (int ni = 0; ni < 4; ++ni) {
        const int gc = nBase + wn + ni * 16 + l16;
        const int g0 = mBase + wm + mi * 16 + rb;
        const float bv = bias[gc];
#pragma unroll
        for (int i = 0; i < 8; ++i) {
          float val = c[mi][ni][i] + bv;
          if (RELU) val = fmaxf(val, 0.0f);
          if (OUTBF16)
            reinterpret_cast<unsigned short*>(Dout)[(size_t)(g0 + i) * N + gc] = f2bfu(val);
          else
            reinterpret_cast<float*>(Dout)[(size_t)(g0 + i) * N + gc] = val;
        }
      }
  } else {
#pragma unroll
    for (int mi = 0; mi < 2; ++mi)
#pragma unroll
      for (int ni = 0; ni < 4; ++ni) {
        const int gc = nBase + wn + ni * 16 + l16;
        const int g0 = mBase + wm + mi * 16 + rb;
        const float bv = bias[gc];
#pragma unroll
        for (int i = 0; i < 8; ++i) {
          if (g0 + i < M) {
            float val = c[mi][ni][i] + bv;
            if (RELU) val = fmaxf(val, 0.0f);
            if (OUTBF16)
              reinterpret_cast<unsigned short*>(Dout)[(size_t)(g0 + i) * N + gc] = f2bfu(val);
            else
              reinterpret_cast<float*>(Dout)[(size_t)(g0 + i) * N + gc] = val;
          }
        }
      }
  }
}

// ---------------- softmax level weights: scale_w[QZ,4] ----------------
__global__ void scale_kernel(const float* __restrict__ pos_emb,
                             const float* __restrict__ wt_w,
                             const float* __restrict__ wt_b,
                             float* __restrict__ scale_w) {
  int p = blockIdx.x * blockDim.x + threadIdx.x;
  if (p >= QZ_) return;
  float lg0 = wt_b[0], lg1 = wt_b[1], lg2 = wt_b[2], lg3 = wt_b[3];
  const float* pe = pos_emb + (size_t)p * C_;
#pragma unroll 4
  for (int c = 0; c < C_; ++c) {
    float e = pe[c];
    const float* w = wt_w + c * 4;
    lg0 += e * w[0]; lg1 += e * w[1]; lg2 += e * w[2]; lg3 += e * w[3];
  }
  float m = fmaxf(fmaxf(lg0, lg1), fmaxf(lg2, lg3));
  float e0 = expf(lg0 - m), e1 = expf(lg1 - m), e2 = expf(lg2 - m), e3 = expf(lg3 - m);
  float inv = 1.0f / (e0 + e1 + e2 + e3);
  float4 o = make_float4(e0 * inv, e1 * inv, e2 * inv, e3 * inv);
  *reinterpret_cast<float4*>(scale_w + (size_t)p * 4) = o;
}

// ---------------- projection + bilinear sampling + residual -> X0 bf16 [Q, Z*C] ----------------
// One wave per (q,z) point; each lane handles 4 contiguous channels (float4 gathers).
__global__ __launch_bounds__(256) void sample_kernel(
    const float* __restrict__ f0, const float* __restrict__ f1,
    const float* __restrict__ f2, const float* __restrict__ f3,
    const float* __restrict__ rp, const float* __restrict__ l2i,
    const float* __restrict__ pos_emb, const float* __restrict__ scale_w,
    unsigned short* __restrict__ X0) {
  int t = blockIdx.x * blockDim.x + threadIdx.x;
  int p = t >> 5;
  int lane = t & 31;
  if (p >= QZ_) return;
  int q = p >> 2, z = p & 3;

  float p0 = rp[((size_t)z * Q_ + q) * 3 + 0];
  float p1 = rp[((size_t)z * Q_ + q) * 3 + 1];
  float p2 = rp[((size_t)z * Q_ + q) * 3 + 2];
  float Xm = p0 * 100.0f - 50.0f;   // pc_range unnormalize
  float Ym = p1 * 100.0f - 50.0f;
  float Zm = p2 * 8.0f - 4.0f;

  float us[NCAM_], vs[NCAM_];
  unsigned vmask = 0;
#pragma unroll
  for (int n = 0; n < NCAM_; ++n) {
    const float* Mx = l2i + n * 16;
    float cx = Mx[0] * Xm + Mx[1] * Ym + Mx[2]  * Zm + Mx[3];
    float cy = Mx[4] * Xm + Mx[5] * Ym + Mx[6]  * Zm + Mx[7];
    float cz = Mx[8] * Xm + Mx[9] * Ym + Mx[10] * Zm + Mx[11];
    float den = fmaxf(cz, 1e-5f);
    float u = cx / (den * 704.0f);
    float v = cy / (den * 256.0f);
    us[n] = u; vs[n] = v;
    if (cz > 1e-5f && u > 0.0f && u < 1.0f && v > 0.0f && v < 1.0f)
      vmask |= (1u << n);
  }

  const int c0 = lane * 4;
  float a0 = 0, a1 = 0, a2 = 0, a3 = 0;
  const float* fb_arr[4] = {f0, f1, f2, f3};
  const int LH[4] = {64, 32, 16, 8};
  const int LW[4] = {176, 88, 44, 22};
  const float4 sw = *reinterpret_cast<const float4*>(scale_w + (size_t)p * 4);
  const float swa[4] = {sw.x, sw.y, sw.z, sw.w};

#pragma unroll
  for (int l = 0; l < 4; ++l) {
    const int Hl = LH[l], Wl = LW[l];
    const float* fb = fb_arr[l];
    float s0 = 0, s1 = 0, s2 = 0, s3 = 0;
    for (int n = 0; n < NCAM_; ++n) {
      if (!((vmask >> n) & 1u)) continue;         // wave-uniform branch
      float pxl = us[n] * Wl - 0.5f;
      float pyl = vs[n] * Hl - 0.5f;
      float fx = floorf(pxl), fy = floorf(pyl);
      int x0i = (int)fx, y0i = (int)fy;
      float wx = pxl - fx, wy = pyl - fy;
      float cw[4] = {(1 - wy) * (1 - wx), (1 - wy) * wx, wy * (1 - wx), wy * wx};
      int xs[4] = {x0i, x0i + 1, x0i, x0i + 1};
      int ys[4] = {y0i, y0i, y0i + 1, y0i + 1};
#pragma unroll
      for (int ci = 0; ci < 4; ++ci) {
        int xi = xs[ci], yi = ys[ci];
        if (yi >= 0 && yi < Hl && xi >= 0 && xi < Wl) {
          const float4 fv = *reinterpret_cast<const float4*>(
              fb + (((size_t)n * Hl + yi) * Wl + xi) * C_ + c0);
          s0 += cw[ci] * fv.x; s1 += cw[ci] * fv.y;
          s2 += cw[ci] * fv.z; s3 += cw[ci] * fv.w;
        }
      }
    }
    float s = swa[l];
    a0 += s0 * s; a1 += s1 * s; a2 += s2 * s; a3 += s3 * s;
  }

  const float4 pe = *reinterpret_cast<const float4*>(pos_emb + (size_t)p * C_ + c0);
  unsigned short h[4] = {f2bfu(a0 + pe.x), f2bfu(a1 + pe.y),
                         f2bfu(a2 + pe.z), f2bfu(a3 + pe.w)};
  __builtin_memcpy(X0 + ((size_t)q * 512 + (size_t)z * C_ + c0), h, 8);
}

extern "C" void kernel_launch(void* const* d_in, const int* in_sizes, int n_in,
                              void* d_out, int out_size, void* d_ws, size_t ws_size,
                              hipStream_t stream) {
  const float* feat0 = (const float*)d_in[0];
  const float* feat1 = (const float*)d_in[1];
  const float* feat2 = (const float*)d_in[2];
  const float* feat3 = (const float*)d_in[3];
  const float* rp    = (const float*)d_in[4];
  const float* l2i   = (const float*)d_in[5];
  const float* pe_w1 = (const float*)d_in[6];
  const float* pe_b1 = (const float*)d_in[7];
  const float* pe_w2 = (const float*)d_in[8];
  const float* pe_b2 = (const float*)d_in[9];
  const float* wt_w  = (const float*)d_in[10];
  const float* wt_b  = (const float*)d_in[11];
  const float* hm_w1 = (const float*)d_in[12];
  const float* hm_b1 = (const float*)d_in[13];
  const float* hm_w2 = (const float*)d_in[14];
  const float* hm_b2 = (const float*)d_in[15];
  const float* hm_w3 = (const float*)d_in[16];
  const float* hm_b3 = (const float*)d_in[17];
  const float* hm_w4 = (const float*)d_in[18];
  const float* hm_b4 = (const float*)d_in[19];
  (void)in_sizes; (void)n_in; (void)out_size; (void)ws_size;

  char* ws = (char*)d_ws;
  size_t off = 0;
  auto alloc = [&](size_t bytes) {
    size_t o = off;
    off += (bytes + 255) & ~(size_t)255;
    return o;
  };
  size_t o_wpe1 = alloc((size_t)64 * 256 * 2);
  size_t o_wpe2 = alloc((size_t)256 * 128 * 2);
  size_t o_wh1  = alloc((size_t)512 * 1024 * 2);
  size_t o_wh2  = alloc((size_t)1024 * 1024 * 2);
  size_t o_wh3  = alloc((size_t)1024 * 1024 * 2);
  size_t o_wh4  = alloc((size_t)1024 * 128 * 2);
  size_t o_pos  = alloc((size_t)QZ_ * 128 * 4);     // pos_emb f32
  size_t o_scl  = alloc((size_t)QZ_ * 4 * 4);       // softmax level weights
  size_t o_bufA = alloc((size_t)20000 * 1024 * 2);  // pe_in / H1 / H3
  size_t o_bufB = alloc((size_t)20000 * 1024 * 2);  // pe_h1 / H2
  size_t o_X0   = alloc((size_t)20000 * 512 * 2);   // fused features bf16

  unsigned short* wpe1 = (unsigned short*)(ws + o_wpe1);
  unsigned short* wpe2 = (unsigned short*)(ws + o_wpe2);
  unsigned short* wh1  = (unsigned short*)(ws + o_wh1);
  unsigned short* wh2  = (unsigned short*)(ws + o_wh2);
  unsigned short* wh3  = (unsigned short*)(ws + o_wh3);
  unsigned short* wh4  = (unsigned short*)(ws + o_wh4);
  float* pos_emb = (float*)(ws + o_pos);
  float* scl     = (float*)(ws + o_scl);
  unsigned short* bufA = (unsigned short*)(ws + o_bufA);
  unsigned short* bufB = (unsigned short*)(ws + o_bufB);
  unsigned short* X0   = (unsigned short*)(ws + o_X0);

  auto cvt = [&](const float* s, unsigned short* d, int K, int N, int Kp) {
    int tot = Kp * N;
    convert_w_kernel<<<(tot + 255) / 256, 256, 0, stream>>>(s, d, K, N, Kp);
  };
  cvt(pe_w1, wpe1, 60, 256, 64);
  cvt(pe_w2, wpe2, 256, 128, 256);
  cvt(hm_w1, wh1, 512, 1024, 512);
  cvt(hm_w2, wh2, 1024, 1024, 1024);
  cvt(hm_w3, wh3, 1024, 1024, 1024);
  cvt(hm_w4, wh4, 1024, 128, 1024);

  pe_feat_kernel<<<(QZ_ + 255) / 256, 256, 0, stream>>>(rp, bufA);

  auto gemm = [&](const unsigned short* A, const unsigned short* W, const float* bias,
                  void* D, int M, int N, int K, bool relu, bool outbf) {
    dim3 grid(N / 128, (M + 127) / 128);
    const __bf16* Ab = reinterpret_cast<const __bf16*>(A);
    const __bf16* Wb = reinterpret_cast<const __bf16*>(W);
    if (relu && outbf)
      gemm_bf16_kernel<true, true><<<grid, 256, 0, stream>>>(Ab, Wb, bias, D, M, N, K);
    else if (relu && !outbf)
      gemm_bf16_kernel<true, false><<<grid, 256, 0, stream>>>(Ab, Wb, bias, D, M, N, K);
    else if (!relu && outbf)
      gemm_bf16_kernel<false, true><<<grid, 256, 0, stream>>>(Ab, Wb, bias, D, M, N, K);
    else
      gemm_bf16_kernel<false, false><<<grid, 256, 0, stream>>>(Ab, Wb, bias, D, M, N, K);
  };

  // PE MLP
  gemm(bufA, wpe1, pe_b1, bufB, QZ_, 256, 64, true, true);
  gemm(bufB, wpe2, pe_b2, pos_emb, QZ_, 128, 256, false, false);

  // level weights + sampling (fuses +pos_emb residual, writes bf16 X0)
  scale_kernel<<<(QZ_ + 255) / 256, 256, 0, stream>>>(pos_emb, wt_w, wt_b, scl);
  sample_kernel<<<(QZ_ * 32) / 256, 256, 0, stream>>>(feat0, feat1, feat2, feat3,
                                                      rp, l2i, pos_emb, scl, X0);

  // head MLP (dominant compute, all WMMA)
  gemm(X0,   wh1, hm_b1, bufA, 20000, 1024, 512,  true,  true);
  gemm(bufA, wh2, hm_b2, bufB, 20000, 1024, 1024, true,  true);
  gemm(bufB, wh3, hm_b3, bufA, 20000, 1024, 1024, true,  true);
  gemm(bufA, wh4, hm_b4, d_out, 20000, 128, 1024, false, false);
}
